// MultiHeadAttentionBlock_81544249082345
// MI455X (gfx1250) — compile-verified, matrix operand order checked by reading, not run on hardware
//
#include <hip/hip_runtime.h>
#include <hip/hip_bf16.h>
#include <math.h>

// ---------------- types ----------------
typedef __bf16 v16bf __attribute__((ext_vector_type(16)));
typedef float  v8f   __attribute__((ext_vector_type(8)));
typedef __bf16 bf16_t;

union AB16 { v16bf v; unsigned u[8]; };

#define D_EMBED 1024
#define S_LEN   2048
#define B_SIZE  4
#define N_HEADS 16
#define D_HEAD  64

__device__ __forceinline__ v8f v8f_zero() {
  v8f z;
#pragma unroll
  for (int i = 0; i < 8; ++i) z[i] = 0.0f;
  return z;
}

// Raw 32-bit LDS offset of a shared-memory element (generic ptr low bits).
__device__ __forceinline__ unsigned lds_off(const void* p) {
  return (unsigned)(uintptr_t)p;
}

// Async global->LDS dword copy (CDNA5). Tracked on ASYNCcnt.
__device__ __forceinline__ void async_copy_b32(unsigned lds_addr,
                                               const void* gaddr) {
  unsigned long long ga = (unsigned long long)(uintptr_t)gaddr;
  asm volatile("global_load_async_to_lds_b32 %0, %1, off"
               :: "v"(lds_addr), "v"(ga) : "memory");
}
__device__ __forceinline__ void wait_async() {
  asm volatile("s_wait_asynccnt 0x0" ::: "memory");
}

// =====================================================================
// Kernel 1: QKV projection GEMM  (x[8192,1024] f32 @ W[1024,1024] + b)
// Output: bf16, layout [B, H, S, Dh].  blockIdx.z selects Q/K/V.
// K staged in 64-deep chunks (16 WMMAs per barrier interval per wave).
// =====================================================================
__global__ __launch_bounds__(256) void qkv_gemm_kernel(
    const float* __restrict__ x,
    const float* __restrict__ wq, const float* __restrict__ bq,
    const float* __restrict__ wk, const float* __restrict__ bk,
    const float* __restrict__ wv, const float* __restrict__ bv,
    bf16_t* __restrict__ qo, bf16_t* __restrict__ ko, bf16_t* __restrict__ vo)
{
  __shared__ __align__(16) bf16_t ldsA[128 * 66];  // [m][k], padded (33 dwords)
  __shared__ __align__(16) bf16_t ldsB[128 * 66];  // [n][k], padded

  const int t = threadIdx.x;
  const int lane = t & 31, wid = t >> 5;
  const int l16 = lane & 15, hh = lane >> 4;
  const int n0 = blockIdx.x * 128, m0 = blockIdx.y * 128;
  const int z = blockIdx.z;
  const float* w    = (z == 0) ? wq : (z == 1) ? wk : wv;
  const float* bias = (z == 0) ? bq : (z == 1) ? bk : bv;
  bf16_t* outp = (z == 0) ? qo : (z == 1) ? ko : vo;

  // K-pair offsets for 16-bit A/B fragment layout (16x32, wave32)
  int koff[8];
#pragma unroll
  for (int j = 0; j < 8; ++j) koff[j] = ((j < 4) ? 0 : 16) + hh * 8 + 2 * (j & 3);

  const int wm = wid >> 2, wn = wid & 3;   // wave tile: 64 rows x 32 cols
  v8f acc[4][2];
#pragma unroll
  for (int i = 0; i < 4; ++i)
#pragma unroll
    for (int jj = 0; jj < 2; ++jj) acc[i][jj] = v8f_zero();

  for (int k0 = 0; k0 < D_EMBED; k0 += 64) {
    // stage A tile (f32 -> bf16 via hw cvt), row-major [128][64]
    for (int e = t; e < 8192; e += 256) {
      int row = e >> 6, kk = e & 63;
      ldsA[row * 66 + kk] = (bf16_t)x[(m0 + row) * D_EMBED + k0 + kk];
    }
    // stage W tile transposed -> [n][k] so packed k-pairs are contiguous
    for (int e = t; e < 8192; e += 256) {
      int kk = e >> 7, nn = e & 127;
      ldsB[nn * 66 + kk] = (bf16_t)w[(k0 + kk) * D_EMBED + n0 + nn];
    }
    __syncthreads();

#pragma unroll
    for (int c = 0; c < 2; ++c) {
      AB16 bfr[2];
#pragma unroll
      for (int jj = 0; jj < 2; ++jj) {
        int ncol = wn * 32 + jj * 16 + l16;
#pragma unroll
        for (int j = 0; j < 8; ++j)
          bfr[jj].u[j] = *(const unsigned*)&ldsB[ncol * 66 + c * 32 + koff[j]];
      }
#pragma unroll
      for (int i = 0; i < 4; ++i) {
        AB16 afr;
        int rowm = wm * 64 + i * 16 + l16;
#pragma unroll
        for (int j = 0; j < 8; ++j)
          afr.u[j] = *(const unsigned*)&ldsA[rowm * 66 + c * 32 + koff[j]];
#pragma unroll
        for (int jj = 0; jj < 2; ++jj)
          acc[i][jj] = __builtin_amdgcn_wmma_f32_16x16x32_bf16(
              false, afr.v, false, bfr[jj].v, (short)0, acc[i][jj], false, false);
      }
    }
    __syncthreads();
  }

  // epilogue: +bias, convert bf16, scatter into [B,H,S,Dh]
#pragma unroll
  for (int jj = 0; jj < 2; ++jj) {
    int n_g = n0 + wn * 32 + jj * 16 + l16;
    float bval = bias[n_g];
    int head = n_g >> 6, dh = n_g & 63;
#pragma unroll
    for (int i = 0; i < 4; ++i) {
#pragma unroll
      for (int r = 0; r < 8; ++r) {
        int m_g = m0 + wm * 64 + i * 16 + r + 8 * hh;
        int bb = m_g >> 11, ss = m_g & (S_LEN - 1);
        outp[((size_t)(bb * N_HEADS + head) * S_LEN + ss) * D_HEAD + dh] =
            (bf16_t)(acc[i][jj][r] + bval);
      }
    }
  }
}

// =====================================================================
// Kernel 2: flash attention. One block per (qblock=128, head, batch).
// 8 waves, each owns 16 query rows. Key blocks of 32.
// K tiles staged with async global->LDS copies (ASYNCcnt path).
// =====================================================================
__global__ __launch_bounds__(256) void attn_kernel(
    const bf16_t* __restrict__ Qw, const bf16_t* __restrict__ Kw,
    const bf16_t* __restrict__ Vw, const int* __restrict__ mask,
    bf16_t* __restrict__ Ow)
{
  __shared__ __align__(16) bf16_t ldsK[32 * 66];       // [key][dh], padded
  __shared__ __align__(16) bf16_t ldsV[64 * 34];       // [dh][key] (transposed)
  __shared__ __align__(16) bf16_t ldsP[8 * 16 * 34];   // per-wave P scratch

  const int t = threadIdx.x, lane = t & 31, wid = t >> 5;
  const int l16 = lane & 15, hh = lane >> 4;
  const int head = blockIdx.y, batch = blockIdx.z;
  const int q0 = blockIdx.x * 128 + wid * 16;
  const size_t bh = (size_t)(batch * N_HEADS + head) * S_LEN;
  const unsigned* Qu = (const unsigned*)(Qw + bh * D_HEAD);  // 32 dwords/row
  const unsigned* Ku = (const unsigned*)(Kw + bh * D_HEAD);
  const unsigned* Vu = (const unsigned*)(Vw + bh * D_HEAD);

  int koff[8];
#pragma unroll
  for (int j = 0; j < 8; ++j) koff[j] = ((j < 4) ? 0 : 16) + hh * 8 + 2 * (j & 3);

  // Q fragments held in registers: A layout 16x32, dh chunks c=0,1
  AB16 qf[2];
#pragma unroll
  for (int c = 0; c < 2; ++c) {
    int row = q0 + l16;
#pragma unroll
    for (int j = 0; j < 8; ++j)
      qf[c].u[j] = Qu[row * 32 + c * 16 + (koff[j] >> 1)];
  }

  float m_r[8], l_r[8];
#pragma unroll
  for (int r = 0; r < 8; ++r) { m_r[r] = -INFINITY; l_r[r] = 0.0f; }
  v8f O[4];
#pragma unroll
  for (int t4 = 0; t4 < 4; ++t4) O[t4] = v8f_zero();

  for (int kb = 0; kb < S_LEN; kb += 32) {
    // stage K block [32 keys][64 dh] via async global->LDS (padded 33 dwords/row)
    for (int e = t; e < 1024; e += 256) {
      int row = e >> 5, c = e & 31;
      async_copy_b32(lds_off(&((unsigned*)ldsK)[row * 33 + c]),
                     Ku + (size_t)(kb + row) * 32 + c);
    }
    // stage V block transposed -> [dh][key] (manual: transpose on the way in)
    for (int e = t; e < 1024; e += 256) {
      int key = e >> 5, c = e & 31;
      unsigned v = Vu[(kb + key) * 32 + c];
      ((unsigned short*)ldsV)[(2 * c) * 34 + key]     = (unsigned short)(v & 0xFFFFu);
      ((unsigned short*)ldsV)[(2 * c + 1) * 34 + key] = (unsigned short)(v >> 16);
    }
    wait_async();
    __syncthreads();

    // S = Q K^T : two 16x16 score tiles, each accumulated over 2 dh chunks
    v8f s[2];
    s[0] = v8f_zero(); s[1] = v8f_zero();
#pragma unroll
    for (int kk2 = 0; kk2 < 2; ++kk2) {
#pragma unroll
      for (int c = 0; c < 2; ++c) {
        AB16 kf;
        int ncol = kk2 * 16 + l16;  // key column within block
#pragma unroll
        for (int j = 0; j < 8; ++j)
          kf.u[j] = ((const unsigned*)ldsK)[ncol * 33 + c * 16 + (koff[j] >> 1)];
        s[kk2] = __builtin_amdgcn_wmma_f32_16x16x32_bf16(
            false, qf[c].v, false, kf.v, (short)0, s[kk2], false, false);
      }
    }
    // scale 1/sqrt(Dh) and mask
#pragma unroll
    for (int kk2 = 0; kk2 < 2; ++kk2) {
#pragma unroll
      for (int r = 0; r < 8; ++r) {
        int qrow = q0 + r + 8 * hh;
        int kcol = kb + kk2 * 16 + l16;
        float sv = s[kk2][r] * 0.125f;
        if (mask[qrow * S_LEN + kcol] == 0) sv = -1e9f;
        s[kk2][r] = sv;
      }
    }
    // online softmax (row reductions across 16-lane halves)
    float p0[8], p1[8], cf[8];
#pragma unroll
    for (int r = 0; r < 8; ++r) {
      float mx = fmaxf(s[0][r], s[1][r]);
      mx = fmaxf(mx, __shfl_xor(mx, 1));
      mx = fmaxf(mx, __shfl_xor(mx, 2));
      mx = fmaxf(mx, __shfl_xor(mx, 4));
      mx = fmaxf(mx, __shfl_xor(mx, 8));
      float newm = fmaxf(m_r[r], mx);
      float c_ = __expf(m_r[r] - newm);
      m_r[r] = newm; cf[r] = c_;
      p0[r] = __expf(s[0][r] - newm);
      p1[r] = __expf(s[1][r] - newm);
      float rs = p0[r] + p1[r];
      rs += __shfl_xor(rs, 1);
      rs += __shfl_xor(rs, 2);
      rs += __shfl_xor(rs, 4);
      rs += __shfl_xor(rs, 8);
      l_r[r] = l_r[r] * c_ + rs;
    }
#pragma unroll
    for (int t4 = 0; t4 < 4; ++t4)
#pragma unroll
      for (int r = 0; r < 8; ++r) O[t4][r] *= cf[r];

    // P: C-layout -> A-fragment layout via per-wave LDS scratch
    bf16_t* myP = &ldsP[wid * 16 * 34];
#pragma unroll
    for (int r = 0; r < 8; ++r) {
      int m = r + 8 * hh;
      myP[m * 34 + l16]      = (bf16_t)p0[r];
      myP[m * 34 + 16 + l16] = (bf16_t)p1[r];
    }
    __syncthreads();
    AB16 pf;
#pragma unroll
    for (int j = 0; j < 8; ++j)
      pf.u[j] = *(const unsigned*)&myP[l16 * 34 + koff[j]];

    // O += P V  (B fragments from transposed V, packed key pairs contiguous)
#pragma unroll
    for (int t4 = 0; t4 < 4; ++t4) {
      AB16 vf;
      int ncol = t4 * 16 + l16;  // dh column
#pragma unroll
      for (int j = 0; j < 8; ++j)
        vf.u[j] = *(const unsigned*)&ldsV[ncol * 34 + koff[j]];
      O[t4] = __builtin_amdgcn_wmma_f32_16x16x32_bf16(
          false, pf.v, false, vf.v, (short)0, O[t4], false, false);
    }
    __syncthreads();
  }

  // normalize and store to [B, S, D] bf16
#pragma unroll
  for (int t4 = 0; t4 < 4; ++t4) {
#pragma unroll
    for (int r = 0; r < 8; ++r) {
      int qrow = q0 + r + 8 * hh;
      int dh = t4 * 16 + l16;
      float val = O[t4][r] / l_r[r];
      Ow[(size_t)(batch * S_LEN + qrow) * D_EMBED + head * D_HEAD + dh] = (bf16_t)val;
    }
  }
}

// =====================================================================
// Kernel 3: output projection. attn(bf16)[8192,1024] @ wo + bo -> f32 out
// A tiles staged with async global->LDS copies.
// =====================================================================
__global__ __launch_bounds__(256) void out_gemm_kernel(
    const bf16_t* __restrict__ a, const float* __restrict__ w,
    const float* __restrict__ bias, float* __restrict__ out)
{
  __shared__ __align__(16) bf16_t ldsA[128 * 66];
  __shared__ __align__(16) bf16_t ldsB[128 * 66];

  const int t = threadIdx.x;
  const int lane = t & 31, wid = t >> 5;
  const int l16 = lane & 15, hh = lane >> 4;
  const int n0 = blockIdx.x * 128, m0 = blockIdx.y * 128;

  int koff[8];
#pragma unroll
  for (int j = 0; j < 8; ++j) koff[j] = ((j < 4) ? 0 : 16) + hh * 8 + 2 * (j & 3);

  const int wm = wid >> 2, wn = wid & 3;
  v8f acc[4][2];
#pragma unroll
  for (int i = 0; i < 4; ++i)
#pragma unroll
    for (int jj = 0; jj < 2; ++jj) acc[i][jj] = v8f_zero();

  const unsigned* au = (const unsigned*)a;

  for (int k0 = 0; k0 < D_EMBED; k0 += 64) {
    // stage A (already bf16): async dword copy, padded stride 33 dwords
    for (int e = t; e < 4096; e += 256) {
      int row = e >> 5, kc = e & 31;
      async_copy_b32(lds_off(&((unsigned*)ldsA)[row * 33 + kc]),
                     au + (size_t)(m0 + row) * (D_EMBED / 2) + (k0 >> 1) + kc);
    }
    // stage W transposed (f32 -> bf16 via hw cvt)
    for (int e = t; e < 8192; e += 256) {
      int kk = e >> 7, nn = e & 127;
      ldsB[nn * 66 + kk] = (bf16_t)w[(k0 + kk) * D_EMBED + n0 + nn];
    }
    wait_async();
    __syncthreads();

#pragma unroll
    for (int c = 0; c < 2; ++c) {
      AB16 bfr[2];
#pragma unroll
      for (int jj = 0; jj < 2; ++jj) {
        int ncol = wn * 32 + jj * 16 + l16;
#pragma unroll
        for (int j = 0; j < 8; ++j)
          bfr[jj].u[j] = *(const unsigned*)&ldsB[ncol * 66 + c * 32 + koff[j]];
      }
#pragma unroll
      for (int i = 0; i < 4; ++i) {
        AB16 afr;
        int rowm = wm * 64 + i * 16 + l16;
#pragma unroll
        for (int j = 0; j < 8; ++j)
          afr.u[j] = *(const unsigned*)&ldsA[rowm * 66 + c * 32 + koff[j]];
#pragma unroll
        for (int jj = 0; jj < 2; ++jj)
          acc[i][jj] = __builtin_amdgcn_wmma_f32_16x16x32_bf16(
              false, afr.v, false, bfr[jj].v, (short)0, acc[i][jj], false, false);
      }
    }
    __syncthreads();
  }

#pragma unroll
  for (int jj = 0; jj < 2; ++jj) {
    int n_g = n0 + wn * 32 + jj * 16 + l16;
    float bval = bias[n_g];
#pragma unroll
    for (int i = 0; i < 4; ++i) {
#pragma unroll
      for (int r = 0; r < 8; ++r) {
        int m_g = m0 + wm * 64 + i * 16 + r + 8 * hh;
        out[(size_t)m_g * D_EMBED + n_g] = acc[i][jj][r] + bval;
      }
    }
  }
}

// =====================================================================
extern "C" void kernel_launch(void* const* d_in, const int* in_sizes, int n_in,
                              void* d_out, int out_size, void* d_ws, size_t ws_size,
                              hipStream_t stream) {
  const float* x  = (const float*)d_in[0];
  const int* mask = (const int*)d_in[1];
  const float* wq = (const float*)d_in[2];
  const float* bq = (const float*)d_in[3];
  const float* wk = (const float*)d_in[4];
  const float* bk = (const float*)d_in[5];
  const float* wv = (const float*)d_in[6];
  const float* bv = (const float*)d_in[7];
  const float* wo = (const float*)d_in[8];
  const float* bo = (const float*)d_in[9];
  float* out = (float*)d_out;

  bf16_t* ws = (bf16_t*)d_ws;
  const size_t elems = (size_t)B_SIZE * S_LEN * D_EMBED;  // 8,388,608
  bf16_t* Qw = ws;
  bf16_t* Kw = Qw + elems;
  bf16_t* Vw = Kw + elems;
  bf16_t* Aw = Vw + elems;   // attention output, [B,S,D] bf16

  dim3 blk(256);
  dim3 g1(D_EMBED / 128, (B_SIZE * S_LEN) / 128, 3);
  qkv_gemm_kernel<<<g1, blk, 0, stream>>>(x, wq, bq, wk, bk, wv, bv, Qw, Kw, Vw);

  dim3 g2(S_LEN / 128, N_HEADS, B_SIZE);
  attn_kernel<<<g2, blk, 0, stream>>>(Qw, Kw, Vw, mask, Aw);

  dim3 g3(D_EMBED / 128, (B_SIZE * S_LEN) / 128, 1);
  out_gemm_kernel<<<g3, blk, 0, stream>>>(Aw, wo, bo, out);
}
